// DownBlock_68109591381036
// MI455X (gfx1250) — matrix operand order repaired and checked by reading, not compiled, not run on hardware
//
#include <hip/hip_runtime.h>

typedef __attribute__((ext_vector_type(16))) _Float16 v16h;
typedef __attribute__((ext_vector_type(8)))  float    v8f;

#define B_      4
#define C_IN    64
#define C_OUT   128
#define H_      256
#define W_      256
#define HO_     128
#define WO_     128
#define NPOS    (HO_ * WO_)        // 16384
#define K2_     9
#define KDIM    (C_IN * K2_)       // 576
#define CHUNKS  (KDIM / 32)        // 18
#define OFFCH   (2 * K2_)          // 18
#define NGRP    16
#define EPS_    1e-5f
#define KP      592                // padded K stride in val LDS tile (32B-aligned rows)

// Offset conv implicit GEMM: taps padded 49 -> 50, K = c_hi*800 + tap*16 + c_lo
#define OFFK2    3200              // 4 * 800
#define OFFCHK2  100               // OFFK2 / 32 (25 chunks per 16-channel slice)

// ---------------------------------------------------------------------------
// Repack main conv weights [C_OUT, C_IN, 3, 3] (f32) into WMMA A-fragment
// order: wp[((mt*CHUNKS + kk)*32 + lane)*16 + slot], K index = tap*64 + c.
// K order used here (B matches): lane<16 slots -> K 0..15; lane>=16 -> K 16..31.
// ---------------------------------------------------------------------------
__global__ void prep_w_main(const float* __restrict__ w, _Float16* __restrict__ wp) {
    int idx  = blockIdx.x * 256 + threadIdx.x;       // < 8*18*32*16 = 73728
    int slot = idx & 15;
    int lane = (idx >> 4) & 31;
    int kk   = (idx >> 9) % CHUNKS;
    int mt   = idx / (512 * CHUNKS);
    int o    = mt * 16 + (lane & 15);
    int Kloc = slot + ((lane & 16) ? 16 : 0);
    int K    = kk * 32 + Kloc;
    int tap  = K >> 6;                               // K = tap*64 + c
    int c    = K & 63;
    wp[idx] = (_Float16)w[(o * C_IN + c) * K2_ + tap];
}

// ---------------------------------------------------------------------------
// Repack offset conv weights [18, C_IN, 7, 7] (padded M=32, taps padded to 50):
// K = c_hi*800 + tap*16 + c_lo  (c = c_hi*16 + c_lo, tap 49 -> zero)
// ---------------------------------------------------------------------------
__global__ void prep_w_off(const float* __restrict__ w, _Float16* __restrict__ wp) {
    int idx  = blockIdx.x * 256 + threadIdx.x;       // < 2*100*32*16 = 102400
    int slot = idx & 15;
    int lane = (idx >> 4) & 31;
    int kc   = (idx >> 9) % OFFCHK2;
    int mt   = idx / (512 * OFFCHK2);
    int o    = mt * 16 + (lane & 15);
    int Kloc = slot + ((lane & 16) ? 16 : 0);
    int K    = kc * 32 + Kloc;                       // 0..3199
    int chi  = K / 800;
    int r    = K - chi * 800;
    int tap  = r >> 4;                               // 0..49
    int clo  = r & 15;
    int c    = chi * 16 + clo;
    float v  = (o < OFFCH && tap < 49) ? w[(o * C_IN + c) * 49 + tap] : 0.0f;
    wp[idx] = (_Float16)v;
}

__global__ void zero_stats(float* s) { s[threadIdx.x] = 0.0f; }   // 128 threads

// ---------------------------------------------------------------------------
// Offset conv as implicit GEMM: M=32 (18 valid), K=3200 (padded), N=65536.
// Block: one (b, oy, half-row of 64 ox). 8 waves = 2 M-tiles x 4 N-tiles.
// LDS patch [ry 0..6][cx 0..135][c_lo 0..15] f16 (30.5 KB), 4 channel passes.
// A chunk of K=32 = 2 consecutive taps x 16 contiguous channels, so each
// lane's B fragment is ONE contiguous 32B v16h in LDS (2x ds_load_b128).
// Inner loop software-pipelined: fragments for chunk k+1 are prefetched
// before the WMMA on chunk k, so waits cover a full iteration of latency.
// ---------------------------------------------------------------------------
__global__ void __launch_bounds__(256) offset_conv(
        const float* __restrict__ x, const _Float16* __restrict__ wp,
        const float* __restrict__ b_off, float* __restrict__ off) {
    __shared__ _Float16 patch[7 * 136 * 16];         // 15232 halfs = 30464 B

    int blk = blockIdx.x;                            // 1024 blocks
    int b   = blk >> 8;
    int r   = blk & 255;
    int oy  = r >> 1;
    int oxb = (r & 1) * 64;
    int gyb = oy * 2 - 3;
    int gxb = oxb * 2 - 3;

    int tid   = threadIdx.x;
    int lane  = tid & 31;
    int w     = tid >> 5;
    int mt    = w >> 2;                              // M tile 0/1
    int nt    = w & 3;                               // N tile 0..3
    int posL  = nt * 16 + (lane & 15);               // 0..63
    int thalf = (lane >> 4) & 1;                     // tap select within chunk

    const _Float16* ap = wp + ((size_t)mt * OFFCHK2 * 32 + lane) * 16;   // +kc*512
    const _Float16* pb = &patch[posL * 32];                              // +*16

    v8f acc = {};
    for (int ch = 0; ch < 4; ++ch) {                 // 16-channel slices
        __syncthreads();
        // stage: cx-fastest => coalesced global reads (128B per wave)
        for (int i = tid; i < 7 * 136 * 16; i += 256) {   // 60 iterations
            int cl = i / 952;                        // c_lo
            int rr = i - cl * 952;
            int ry = rr / 136;
            int cx = rr - ry * 136;
            int gy = gyb + ry;
            int gx = gxb + cx;
            float v = 0.0f;
            if (cx < 133 && gy >= 0 && gy < H_ && gx >= 0 && gx < W_)
                v = x[((b * C_IN + ch * 16 + cl) * H_ + gy) * W_ + gx];
            patch[(ry * 136 + cx) * 16 + cl] = (_Float16)v;
        }
        __syncthreads();

        const _Float16* apc = ap + ch * 25 * 512;
        // prologue: fragments for kloc = 0
        int t0 = thalf;                              // kloc*2 + thalf, kloc=0
        int ky0 = t0 / 7, kx0 = t0 - ky0 * 7;
        v16h a  = *(const v16h*)(apc);
        v16h bv = *(const v16h*)(pb + (ky0 * 136 + kx0) * 16);
        for (int kloc = 0; kloc < 25; ++kloc) {
            int kn = (kloc + 1 < 25) ? kloc + 1 : kloc;    // clamp (stay in slice)
            int t  = kn * 2 + thalf;
            if (t > 48) t = 48;                      // tap 49: weights are zero
            int ky = t / 7;
            int kx = t - ky * 7;
            v16h an = *(const v16h*)(apc + kn * 512);
            v16h bn = *(const v16h*)(pb + (ky * 136 + kx) * 16);
            acc = __builtin_amdgcn_wmma_f32_16x16x32_f16(
                    false, a, false, bv, (short)0, acc, false, false);
            a = an;
            bv = bn;
        }
    }
    int ox    = oxb + posL;
    int Mbase = (lane & 16) ? 8 : 0;
#pragma unroll
    for (int rI = 0; rI < 8; ++rI) {
        int o = mt * 16 + Mbase + rI;
        if (o < OFFCH)
            off[((b * OFFCH + o) * HO_ + oy) * WO_ + ox] = acc[rI] + b_off[o];
    }
}

// ---------------------------------------------------------------------------
// Deformable conv GEMM + bias + GroupNorm partial stats.
// Block: (b, tile of 16 consecutive output positions). 8 waves = 8 M-tiles.
// ---------------------------------------------------------------------------
__global__ void __launch_bounds__(256) deform_wmma(
        const float* __restrict__ x, const float* __restrict__ off,
        const _Float16* __restrict__ wp, const float* __restrict__ bias,
        float* __restrict__ out, float* __restrict__ stats) {
    __shared__ _Float16 valLds[16 * KP];             // 18944 B, [pos][K]
    __shared__ float    cw[K2_ * 16 * 4];            // bilinear corner weights
    __shared__ int      cyx[K2_ * 16 * 2];           // corner y0/x0

    int blk = blockIdx.x;                            // 4096 blocks
    int b   = blk >> 10;
    int tp  = blk & 1023;
    int p0  = tp * 16;
    int oy  = p0 >> 7;
    int ox0 = p0 & 127;
    int tid = threadIdx.x;

    // Phase 0: per (pos,tap) sampling coordinates from predicted offsets
    if (tid < 144) {
        int pos = tid & 15;
        int tap = tid >> 4;
        int ky  = tap / 3;
        int kx  = tap - ky * 3;
        int p   = p0 + pos;
        float dy = off[(b * OFFCH + tap * 2 + 0) * NPOS + p];
        float dx = off[(b * OFFCH + tap * 2 + 1) * NPOS + p];
        float yf = (float)(oy * 2 - 1 + ky) + dy;
        float xf = (float)((ox0 + pos) * 2 - 1 + kx) + dx;
        float y0f = floorf(yf), x0f = floorf(xf);
        float wy1 = yf - y0f, wx1 = xf - x0f;
        float wy0 = 1.f - wy1, wx0 = 1.f - wx1;
        cw[tid * 4 + 0] = wy0 * wx0;
        cw[tid * 4 + 1] = wy0 * wx1;
        cw[tid * 4 + 2] = wy1 * wx0;
        cw[tid * 4 + 3] = wy1 * wx1;
        cyx[tid * 2 + 0] = (int)y0f;
        cyx[tid * 2 + 1] = (int)x0f;
    }
    __syncthreads();

    // Phase 1: irregular bilinear gather -> f16 im2col tile [pos][tap*64+c]
    for (int e = tid; e < 16 * KDIM; e += 256) {     // 36 iterations, all lanes
        int pos  = e & 15;
        int kIdx = e >> 4;                           // tap*64 + c
        int tap  = kIdx >> 6;
        int c    = kIdx & 63;
        int pair = tap * 16 + pos;
        int y0 = cyx[pair * 2 + 0];
        int x0 = cyx[pair * 2 + 1];
        float w00 = cw[pair * 4 + 0], w01 = cw[pair * 4 + 1];
        float w10 = cw[pair * 4 + 2], w11 = cw[pair * 4 + 3];
        const float* img = x + (size_t)(b * C_IN + c) * (H_ * W_);
        int base = y0 * W_ + x0;
        bool yv0 = (y0 >= 0) && (y0 < H_);
        bool yv1 = (y0 + 1 >= 0) && (y0 + 1 < H_);
        bool xv0 = (x0 >= 0) && (x0 < W_);
        bool xv1 = (x0 + 1 >= 0) && (x0 + 1 < W_);
        float v = 0.0f;
        if (yv0 && xv0) v += w00 * img[base];
        if (yv0 && xv1) v += w01 * img[base + 1];
        if (yv1 && xv0) v += w10 * img[base + W_];
        if (yv1 && xv1) v += w11 * img[base + W_ + 1];
        valLds[pos * KP + kIdx] = (_Float16)v;
    }
    __syncthreads();

    // Phase 2: WMMA, software-pipelined. Wave w -> channels [16w, 16w+16).
    int lane  = tid & 31;
    int w     = tid >> 5;
    int posc  = lane & 15;
    int khalf = lane & 16;
    const _Float16* ap = wp + ((size_t)w * CHUNKS * 32 + lane) * 16;   // +kk*512
    const _Float16* bp = &valLds[posc * KP + khalf];                   // +kk*32
    v16h a  = *(const v16h*)(ap);
    v16h bv = *(const v16h*)(bp);
    v8f acc = {};
    for (int kk = 0; kk < CHUNKS; ++kk) {
        int kn = (kk + 1 < CHUNKS) ? kk + 1 : kk;
        v16h an = *(const v16h*)(ap + kn * 512);
        v16h bn = *(const v16h*)(bp + kn * 32);
        acc = __builtin_amdgcn_wmma_f32_16x16x32_f16(
                false, a, false, bv, (short)0, acc, false, false);
        a = an;
        bv = bn;
    }

    // Epilogue: bias, store pre-GN output, accumulate group stats.
    int Mbase = (lane & 16) ? 8 : 0;                 // each lane: one GN group
    float s1 = 0.f, s2 = 0.f;
#pragma unroll
    for (int rI = 0; rI < 8; ++rI) {
        int o   = w * 16 + Mbase + rI;
        float y = acc[rI] + bias[o];
        out[(b * C_OUT + o) * NPOS + p0 + posc] = y;
        s1 += y;
        s2 += y * y;
    }
#pragma unroll
    for (int m = 1; m < 16; m <<= 1) {               // reduce within each half-wave
        s1 += __shfl_xor(s1, m, 32);
        s2 += __shfl_xor(s2, m, 32);
    }
    if ((lane & 15) == 0) {
        int g = 2 * w + ((lane & 16) ? 1 : 0);
        atomicAdd(&stats[(b * NGRP + g) * 2 + 0], s1);
        atomicAdd(&stats[(b * NGRP + g) * 2 + 1], s2);
    }
}

// ---------------------------------------------------------------------------
// GroupNorm finalize (in place over d_out).
// ---------------------------------------------------------------------------
__global__ void gn_finalize(float* __restrict__ out, const float* __restrict__ stats,
                            const float* __restrict__ gamma, const float* __restrict__ beta) {
    int idx = blockIdx.x * 256 + threadIdx.x;        // < 2^23
    int c   = (idx >> 14) & 127;
    int b   = idx >> 21;
    int g   = c >> 3;
    float n   = (float)(8 * NPOS);
    float s1  = stats[(b * NGRP + g) * 2 + 0];
    float s2  = stats[(b * NGRP + g) * 2 + 1];
    float mu  = s1 / n;
    float var = s2 / n - mu * mu;
    float inv = rsqrtf(var + EPS_);
    float y   = out[idx];
    out[idx]  = (y - mu) * inv * gamma[c] + beta[c];
}

// ---------------------------------------------------------------------------
extern "C" void kernel_launch(void* const* d_in, const int* in_sizes, int n_in,
                              void* d_out, int out_size, void* d_ws, size_t ws_size,
                              hipStream_t stream) {
    (void)in_sizes; (void)n_in; (void)out_size; (void)ws_size;
    const float* x      = (const float*)d_in[0];
    const float* w_off  = (const float*)d_in[1];
    const float* b_off  = (const float*)d_in[2];
    const float* weight = (const float*)d_in[3];
    const float* bias   = (const float*)d_in[4];
    const float* gamma  = (const float*)d_in[5];
    const float* beta   = (const float*)d_in[6];

    char*      ws     = (char*)d_ws;
    float*     off_ws = (float*)ws;                      // 4*18*16384*4  = 4,718,592 B
    float*     stats  = (float*)(ws + 4718592);          // 128*4         = 512 B
    _Float16*  wpMain = (_Float16*)(ws + 4719104);       // 73728*2       = 147,456 B
    _Float16*  wpOff  = (_Float16*)(ws + 4866560);       // 102400*2      = 204,800 B

    prep_w_main<<<288, 256, 0, stream>>>(weight, wpMain);
    prep_w_off <<<400, 256, 0, stream>>>(w_off, wpOff);
    zero_stats <<<1, 128, 0, stream>>>(stats);
    offset_conv<<<1024, 256, 0, stream>>>(x, wpOff, b_off, off_ws);
    deform_wmma<<<4096, 256, 0, stream>>>(x, off_ws, wpMain, bias, (float*)d_out, stats);
    gn_finalize<<<32768, 256, 0, stream>>>((float*)d_out, stats, gamma, beta);
}
